// LRMF_71854802862666
// MI455X (gfx1250) — matrix operand order repaired and checked by reading, not compiled, not run on hardware
//
#include <hip/hip_runtime.h>

typedef __attribute__((ext_vector_type(16))) _Float16 v16h;
typedef __attribute__((ext_vector_type(8)))  float    v8f;

#define U_DIM 4096
#define S_DIM 8192
#define D_DIM 32
#define N_REG 128
#define GA_W  0.1f

// ---------------- workspace layout (bytes) ----------------
// A_packed : U_DIM*32 halves                    @ 0        (262144 B)
// B_packed : S_DIM*32 halves                    @ 262144   (524288 B)
// reg_sums : N_REG*D_DIM floats                 @ 786432   (16384 B)
// reg_cnts : N_REG floats                       @ 802816   (512 B)
#define WS_A    0
#define WS_B    262144
#define WS_SUM  786432
#define WS_CNT  802816

__global__ void zero_ws_kernel(float* out, float* sums, float* cnts) {
    int t = blockIdx.x * blockDim.x + threadIdx.x;
    if (t == 0) out[0] = 0.0f;
    if (t < N_REG * D_DIM) sums[t] = 0.0f;
    if (t < N_REG) cnts[t] = 0.0f;
}

// Pack user_emb (f32, row-major [U,32]) into per-lane WMMA A-fragment order.
// 16-bit A 16x32 (MxK), wave32: lane L -> M = L%16, g = L/16;
// halves 0..7  -> K = 8g + h ; halves 8..15 -> K = 16 + 8g + (h-8).
__global__ void pack_a_kernel(const float* __restrict__ ue, _Float16* __restrict__ ap) {
    int t = blockIdx.x * blockDim.x + threadIdx.x;          // tile*32 + lane
    if (t >= (U_DIM / 16) * 32) return;
    int tile = t >> 5, lane = t & 31;
    int m = lane & 15, g = lane >> 4;
    const float* src = ue + (size_t)(tile * 16 + m) * D_DIM;
    _Float16* dst = ap + (size_t)t * 16;
#pragma unroll
    for (int h = 0; h < 8; ++h) dst[h] = (_Float16)src[8 * g + h];
#pragma unroll
    for (int h = 0; h < 8; ++h) dst[8 + h] = (_Float16)src[16 + 8 * g + h];
}

// Pack item_emb into B-fragment order. B is KxN with B[k][n] = item_emb[j0+n][k].
// 16-bit B 32x16, wave32: lane L -> N = L%16, kg = L/16;
// halves 0..7 -> K = 8kg + h ; halves 8..15 -> K = 16 + 8kg + (h-8).
__global__ void pack_b_kernel(const float* __restrict__ ie, _Float16* __restrict__ bp) {
    int t = blockIdx.x * blockDim.x + threadIdx.x;          // tile*32 + lane
    if (t >= (S_DIM / 16) * 32) return;
    int tile = t >> 5, lane = t & 31;
    int n = lane & 15, kg = lane >> 4;
    const float* src = ie + (size_t)(tile * 16 + n) * D_DIM;
    _Float16* dst = bp + (size_t)t * 16;
#pragma unroll
    for (int h = 0; h < 8; ++h) dst[h] = (_Float16)src[8 * kg + h];
#pragma unroll
    for (int h = 0; h < 8; ++h) dst[8 + h] = (_Float16)src[16 + 8 * kg + h];
}

// Main fused kernel: 1 block per 16-row i-tile, 8 waves, each wave strides j-tiles.
// C/D layout: lane L, vgpr r -> M = r + 8*(L/16), N = L%16.
__global__ __launch_bounds__(256) void qos_mse_kernel(
        const _Float16* __restrict__ ap, const _Float16* __restrict__ bp,
        const int* __restrict__ mask, const float* __restrict__ qos,
        const float* __restrict__ w, float* __restrict__ out) {
    const int lane = threadIdx.x & 31;
    const int wave = threadIdx.x >> 5;
    const int ti = blockIdx.x;
    const int i0 = ti * 16;
    const int n = lane & 15;
    const int mbase = (lane >> 4) * 8;

    // A fragment: reused for all j-tiles handled by this wave.
    v16h a = *(const v16h*)(ap + ((size_t)ti * 32 + lane) * 16);

    float acc = 0.0f;
    for (int tj = wave; tj < S_DIM / 16; tj += 8) {
        v16h b = *(const v16h*)(bp + ((size_t)tj * 32 + lane) * 16);
        v8f c = {};
        c = __builtin_amdgcn_wmma_f32_16x16x32_f16(
                /*neg_a=*/false, a, /*neg_b=*/false, b,
                /*c_mod=*/(short)0, c, /*reuse_a=*/false, /*reuse_b=*/false);
        const int j = tj * 16 + n;
#pragma unroll
        for (int r = 0; r < 8; ++r) {
            const size_t idx = (size_t)(i0 + mbase + r) * S_DIM + j;
            float pre = c[r] * (float)mask[idx];
            float d = pre - qos[idx];
            acc = fmaf(w[idx] * d, d, acc);
        }
    }

    // wave32 reduction, then cross-wave via LDS, one atomic per block
#pragma unroll
    for (int off = 16; off > 0; off >>= 1) acc += __shfl_down(acc, off, 32);
    __shared__ float partial[8];
    if (lane == 0) partial[wave] = acc;
    __syncthreads();
    if (threadIdx.x == 0) {
        float s = 0.0f;
#pragma unroll
        for (int k = 0; k < 8; ++k) s += partial[k];
        atomicAdd(out, s);
    }
}

__global__ void region_accum_kernel(const float* __restrict__ ue,
                                    const int* __restrict__ ridx,
                                    float* __restrict__ sums,
                                    float* __restrict__ cnts) {
    int t = blockIdx.x * blockDim.x + threadIdx.x;
    if (t >= U_DIM * D_DIM) return;
    int u = t >> 5, d = t & 31;
    int r = ridx[u];
    atomicAdd(&sums[r * D_DIM + d], ue[t]);
    if (d == 0) atomicAdd(&cnts[r], 1.0f);
}

__global__ __launch_bounds__(256) void region_final_kernel(
        const float* __restrict__ ue, const int* __restrict__ ridx,
        const float* __restrict__ sums, const float* __restrict__ cnts,
        float* __restrict__ out) {
    int t = blockIdx.x * blockDim.x + threadIdx.x;
    float val = 0.0f;
    if (t < U_DIM * D_DIM) {
        int u = t >> 5, d = t & 31;
        int r = ridx[u];
        float cnt = cnts[r];
        if (cnt > 1.0f) {
            float x = ue[t];
            float loo = (sums[r * D_DIM + d] - x) / fmaxf(cnt - 1.0f, 1.0f);
            val = fabsf(x - loo);
        }
    }
    __shared__ float red[256];
    red[threadIdx.x] = val;
    __syncthreads();
    for (int s = 128; s > 0; s >>= 1) {
        if (threadIdx.x < s) red[threadIdx.x] += red[threadIdx.x + s];
        __syncthreads();
    }
    if (threadIdx.x == 0) atomicAdd(out, GA_W * red[0]);
}

extern "C" void kernel_launch(void* const* d_in, const int* in_sizes, int n_in,
                              void* d_out, int out_size, void* d_ws, size_t ws_size,
                              hipStream_t stream) {
    const float* user_emb = (const float*)d_in[0];   // [U,32] f32
    const float* item_emb = (const float*)d_in[1];   // [S,32] f32
    const int*   mask     = (const int*)d_in[2];     // [U,S] i32
    const float* qos      = (const float*)d_in[3];   // [U,S] f32
    const float* lossw    = (const float*)d_in[4];   // [U,S] f32
    const int*   ridx     = (const int*)d_in[5];     // [U] i32
    float* out = (float*)d_out;

    char* ws = (char*)d_ws;
    _Float16* a_packed = (_Float16*)(ws + WS_A);
    _Float16* b_packed = (_Float16*)(ws + WS_B);
    float* reg_sums = (float*)(ws + WS_SUM);
    float* reg_cnts = (float*)(ws + WS_CNT);

    // 1) zero accumulators
    zero_ws_kernel<<<(N_REG * D_DIM + 255) / 256, 256, 0, stream>>>(out, reg_sums, reg_cnts);

    // 2) pack f32 -> f16 fragments (tiny: 1.5 MB total)
    pack_a_kernel<<<((U_DIM / 16) * 32 + 255) / 256, 256, 0, stream>>>(user_emb, a_packed);
    pack_b_kernel<<<((S_DIM / 16) * 32 + 255) / 256, 256, 0, stream>>>(item_emb, b_packed);

    // 3) region segment sums
    region_accum_kernel<<<(U_DIM * D_DIM + 255) / 256, 256, 0, stream>>>(
        user_emb, ridx, reg_sums, reg_cnts);

    // 4) fused WMMA GEMM + masked weighted MSE (bandwidth-bound main pass)
    qos_mse_kernel<<<U_DIM / 16, 256, 0, stream>>>(a_packed, b_packed, mask, qos, lossw, out);

    // 5) leave-one-out region loss
    region_final_kernel<<<(U_DIM * D_DIM + 255) / 256, 256, 0, stream>>>(
        user_emb, ridx, reg_sums, reg_cnts, out);
}
